// HICFModel_26671746908860
// MI455X (gfx1250) — compile-verified, matrix operand order checked by reading, not run on hardware
//
#include <hip/hip_runtime.h>
#include <hip/hip_fp16.h>

typedef __attribute__((ext_vector_type(16))) _Float16 v16h;
typedef __attribute__((ext_vector_type(2)))  _Float16 v2h;
typedef __attribute__((ext_vector_type(8)))  float    v8f;

#define N_NODES 200000
#define EMB 64
#define E_ADJ 1600000
#define E_TR 65536
#define NUM_NEG 16
#define F_EPS 1e-7f
#define F_MIN_NORM 1e-15f
#define F_MAX_SQ 50.0f
#define F_MARGIN 0.1f

__device__ __forceinline__ float wave_sum32(float v) {
#pragma unroll
    for (int m = 16; m >= 1; m >>= 1) v += __shfl_xor(v, m, 32);
    return v;
}

// ---------------- zero / axpy helpers ----------------
__global__ void k_zero(float* __restrict__ p, long n) {
    long i = blockIdx.x * (long)blockDim.x + threadIdx.x;
    if (i < n) p[i] = 0.0f;
}

__global__ void k_axpy(float* __restrict__ acc, const float* __restrict__ h, long n) {
    long i = blockIdx.x * (long)blockDim.x + threadIdx.x;
    if (i < n) acc[i] += h[i];
}

// ---------------- logmap0: weight -> tangent ----------------
// one wave (32 lanes) per node, 2 dims per lane
__global__ void k_logmap(const float* __restrict__ W, float* __restrict__ out) {
    int wid  = (blockIdx.x * blockDim.x + threadIdx.x) >> 5;
    int lane = threadIdx.x & 31;
    if (wid >= N_NODES) return;
    const float* w = W + (size_t)wid * EMB;
    float2 v = *(const float2*)(w + 2 * lane);
    float s = v.x * v.x + v.y * v.y;
    if (lane == 0) s = v.y * v.y;              // exclude dim 0 from y-norm
    s = wave_sum32(s);
    float x0    = __shfl(v.x, 0, 32);
    float ynorm = fmaxf(sqrtf(s), F_MIN_NORM);
    float r     = acoshf(fmaxf(x0, 1.0f + F_EPS)) / ynorm;   // sqrtK == 1
    float2 o;
    o.x = (lane == 0) ? 0.0f : (v.x * r);
    o.y = v.y * r;
    *(float2*)(out + (size_t)wid * EMB + 2 * lane) = o;
}

// ---------------- SpMM: hout[row] += val * hin[col] ----------------
// one wave per 8 consecutive (sorted-row) edges, 2 dims per lane;
// register accumulation across equal rows, atomic flush on row change.
__global__ void k_spmm(const int* __restrict__ row, const int* __restrict__ col,
                       const float* __restrict__ val, const float* __restrict__ hin,
                       float* __restrict__ hout) {
    const int EPW = 8;
    long wid  = (blockIdx.x * (long)blockDim.x + threadIdx.x) >> 5;
    int  lane = threadIdx.x & 31;
    long e0   = wid * EPW;
    if (e0 >= E_ADJ) return;
    float ax = 0.0f, ay = 0.0f;
    int prev = -1;
#pragma unroll
    for (int k = 0; k < EPW; k++) {
        long e = e0 + k;
        if (e < E_ADJ) {
            int   r = row[e];
            int   c = col[e];
            float s = val[e];
            float2 hv = *(const float2*)(hin + (size_t)c * EMB + 2 * lane);
            if (r != prev) {
                if (prev >= 0) {
                    atomicAdd(hout + (size_t)prev * EMB + 2 * lane,     ax);
                    atomicAdd(hout + (size_t)prev * EMB + 2 * lane + 1, ay);
                }
                ax = 0.0f; ay = 0.0f; prev = r;
            }
            ax += s * hv.x;
            ay += s * hv.y;
        }
    }
    if (prev >= 0) {
        atomicAdd(hout + (size_t)prev * EMB + 2 * lane,     ax);
        atomicAdd(hout + (size_t)prev * EMB + 2 * lane + 1, ay);
    }
}

// ---------------- expmap0 + proj: acc -> final embedding (f32 + f16) ----------------
__global__ void k_final(const float* __restrict__ acc, float* __restrict__ hf32,
                        _Float16* __restrict__ hf16) {
    int wid  = (blockIdx.x * blockDim.x + threadIdx.x) >> 5;
    int lane = threadIdx.x & 31;
    if (wid >= N_NODES) return;
    float2 u = *(const float2*)(acc + (size_t)wid * EMB + 2 * lane);
    float s = u.x * u.x + u.y * u.y;
    if (lane == 0) s = u.y * u.y;              // dims 1..63 only
    s = wave_sum32(s);
    float xn = fmaxf(sqrtf(s), F_MIN_NORM);
    float kk = sinhf(xn) / xn;
    float rx = (lane == 0) ? 0.0f : u.x * kk;
    float ry = u.y * kk;
    // proj: first = sqrt(1 + ||rest||^2)
    float s2 = wave_sum32(rx * rx + ry * ry);
    float first = sqrtf(1.0f + s2);
    float ox = (lane == 0) ? first : rx;
    float2 o; o.x = ox; o.y = ry;
    *(float2*)(hf32 + (size_t)wid * EMB + 2 * lane) = o;
    v2h h2; h2.x = (_Float16)ox; h2.y = (_Float16)ry;
    *(v2h*)(hf16 + (size_t)wid * EMB + 2 * lane) = h2;
}

// ---------------- loss: 1 wave per 16 (anchor,pos) pairs ----------------
// Gram(16x16) of A(16x64) @ P^T(64x16) via two v_wmma_f32_16x16x32_f16;
// for both A and B fragments, lane l loads row (l%16), K-window (l/16)*16.
__global__ void k_loss(const float* __restrict__ hf32, const _Float16* __restrict__ hf16,
                       const int* __restrict__ anchor, const int* __restrict__ pos,
                       const int* __restrict__ neg, float* __restrict__ out) {
    __shared__ float tile[256];
    int l    = threadIdx.x;        // 0..31
    int base = blockIdx.x * 16;
    int m    = l & 15;
    int half = l >> 4;             // 0 or 1 -> K window
    int aidx = anchor[base + m];
    int pidx = pos[base + m];

    const _Float16* arow = hf16 + (size_t)aidx * EMB;
    const _Float16* prow = hf16 + (size_t)pidx * EMB;
    v16h a0 = *(const v16h*)(arow + half * 16);        // K 0..31 fragment
    v16h a1 = *(const v16h*)(arow + 32 + half * 16);   // K 32..63 fragment
    v16h b0 = *(const v16h*)(prow + half * 16);
    v16h b1 = *(const v16h*)(prow + 32 + half * 16);

    v8f c = {};
    c = __builtin_amdgcn_wmma_f32_16x16x32_f16(false, a0, false, b0, (short)0, c, false, false);
    c = __builtin_amdgcn_wmma_f32_16x16x32_f16(false, a1, false, b1, (short)0, c, false, false);

    int mb = (l < 16) ? 0 : 8;     // C/D layout: vgpr r -> M = mb + r, N = l%16
#pragma unroll
    for (int r = 0; r < 8; r++) tile[(mb + r) * 16 + m] = c[r];
    __syncthreads();
    float dot_pos = tile[m * 16 + m];

    // hard-negative argmin: lane pair (l, l^16) shares pair m; each does 8 negatives
    const float* pr32 = hf32 + (size_t)pidx * EMB;
    float best  = 3.4e38f;
    int   bestj = 1 << 30;
    for (int t = 0; t < 8; t++) {
        int j    = half + 2 * t;
        int nidx = neg[(size_t)(base + m) * NUM_NEG + j];
        const float* nr = hf32 + (size_t)nidx * EMB;
        float d2 = 0.0f;
#pragma unroll
        for (int k = 0; k < EMB; k += 4) {
            float4 nv = *(const float4*)(nr + k);
            float4 pv = *(const float4*)(pr32 + k);
            float dx = nv.x - pv.x, dy = nv.y - pv.y;
            float dz = nv.z - pv.z, dw = nv.w - pv.w;
            d2 += dx * dx + dy * dy + dz * dz + dw * dw;
        }
        if (d2 < best || (d2 == best && j < bestj)) { best = d2; bestj = j; }
    }
    float ob = __shfl_xor(best, 16, 32);
    int   oj = __shfl_xor(bestj, 16, 32);
    if (ob < best || (ob == best && oj < bestj)) { best = ob; bestj = oj; }
    int hn = neg[(size_t)(base + m) * NUM_NEG + bestj];

    // Gram of A vs hard negatives
    const _Float16* nrow = hf16 + (size_t)hn * EMB;
    v16h n0 = *(const v16h*)(nrow + half * 16);
    v16h n1 = *(const v16h*)(nrow + 32 + half * 16);
    v8f c2 = {};
    c2 = __builtin_amdgcn_wmma_f32_16x16x32_f16(false, a0, false, n0, (short)0, c2, false, false);
    c2 = __builtin_amdgcn_wmma_f32_16x16x32_f16(false, a1, false, n1, (short)0, c2, false, false);
    __syncthreads();
#pragma unroll
    for (int r = 0; r < 8; r++) tile[(mb + r) * 16 + m] = c2[r];
    __syncthreads();

    float contrib = 0.0f;
    if (l < 16) {
        float dot_neg = tile[m * 16 + m];
        float u0  = hf32[(size_t)aidx * EMB];
        float i0  = hf32[(size_t)pidx * EMB];
        float nn0 = hf32[(size_t)hn * EMB];
        float mink_p = dot_pos - 2.0f * u0 * i0;
        float acp = acoshf(fmaxf(-mink_p, 1.0f + F_EPS));
        float pos_sq = fminf(acp * acp, F_MAX_SQ);
        float mink_n = dot_neg - 2.0f * u0 * nn0;
        float acn = acoshf(fmaxf(-mink_n, 1.0f + F_EPS));
        float neg_sq = fminf(acn * acn, F_MAX_SQ);
        float score = (1.0f - mink_p - u0 - i0) / (u0 * i0);
        float w = 1.0f / (1.0f + expf(score));      // sigmoid(-score)
        contrib = fmaxf(pos_sq - neg_sq + F_MARGIN * w, 0.0f);
    }
    contrib = wave_sum32(contrib);
    if (l == 0) atomicAdd(out, contrib);
}

extern "C" void kernel_launch(void* const* d_in, const int* in_sizes, int n_in,
                              void* d_out, int out_size, void* d_ws, size_t ws_size,
                              hipStream_t stream) {
    const float* W      = (const float*)d_in[0];
    const float* vals   = (const float*)d_in[1];
    const int*   row    = (const int*)d_in[2];
    const int*   col    = (const int*)d_in[3];
    const int*   anchor = (const int*)d_in[4];
    const int*   pos    = (const int*)d_in[5];
    const int*   neg    = (const int*)d_in[6];
    float* out = (float*)d_out;

    char* ws = (char*)d_ws;
    const size_t SZ = (size_t)N_NODES * EMB * sizeof(float);   // 51.2 MB, 256B aligned
    float*     p0   = (float*)(ws);
    float*     p1   = (float*)(ws + SZ);
    float*     acc  = (float*)(ws + 2 * SZ);
    _Float16*  hf16 = (_Float16*)(ws + 3 * SZ);

    const long NE = (long)N_NODES * EMB;                       // 12.8M
    const int  nodeBlocks = (N_NODES + 7) / 8;                 // 8 nodes (waves) per 256-thr block

    k_zero<<<(int)((NE + 255) / 256), 256, 0, stream>>>(acc, NE);
    k_logmap<<<nodeBlocks, 256, 0, stream>>>(W, p0);

    float* cur = p0;
    float* nxt = p1;
    for (int layer = 0; layer < 3; layer++) {
        k_zero<<<(int)((NE + 255) / 256), 256, 0, stream>>>(nxt, NE);
        long waves = (E_ADJ + 7) / 8;                          // 200000 waves
        k_spmm<<<(int)((waves * 32 + 255) / 256), 256, 0, stream>>>(row, col, vals, cur, nxt);
        k_axpy<<<(int)((NE + 255) / 256), 256, 0, stream>>>(acc, nxt, NE);
        float* t = cur; cur = nxt; nxt = t;
    }
    // after 3 swaps: cur==p1, nxt==p0 (free) -> final f32 into nxt
    k_final<<<nodeBlocks, 256, 0, stream>>>(acc, nxt, hf16);

    k_zero<<<1, 1, 0, stream>>>(out, 1);
    k_loss<<<E_TR / 16, 32, 0, stream>>>(nxt, hf16, anchor, pos, neg, out);
}
